// NeuralDecisionForest_49538152792252
// MI455X (gfx1250) — matrix operand order repaired and checked by reading, not compiled, not run on hardware
//
#include <hip/hip_runtime.h>
#include <hip/hip_bf16.h>

// ---------------------------------------------------------------------------
// NeuralDecisionForest for MI455X (gfx1250, wave32, WMMA 16x16x32 f16)
//
// Pipeline:
//   prep   : f32->f16 conversion of x/W1/W2/Wd (Wd padded to 64x256x64),
//            bd padded to 64x256, leaf_preds = sigmoid(leaf_logits)
//   mlp1   : h1 = relu(x @ W1^T + b1)          (8192x128, f16, WMMA)
//   mlp2   : feats = relu(h1 @ W2^T + b2)      (8192x64,  f16, WMMA)
//   forest : fused  d = sigmoid(feats @ Wd^T + bd)  [never materialized],
//            per-level products, 256-leaf contraction, per-tree output.
//            Each wave: 1 tree x 32 batch rows (2 sub-tiles share every Wd
//            A-fragment -> halves Wd L2 traffic vs 1 sub-tile).
//   reduce : out[b] = mean over 64 trees
// ---------------------------------------------------------------------------

typedef __attribute__((ext_vector_type(16))) _Float16 v16h;
typedef __attribute__((ext_vector_type(8)))  _Float16 v8h;
typedef __attribute__((ext_vector_type(8)))  float    v8f;

#define BATCH   8192
#define IND     256
#define H1DIM   128
#define H2DIM   64
#define NTREES  64
#define NNODES  255   // padded to 256
#define NLEAF   256

// ---------------- fragment loaders (CDNA5 16x16x32 f16 layouts) ------------
// A 16x32 (MxK): lanes 0-15 = rows, lanes 16-31 same rows;
// per-lane halfs: K = k0 + 8*hi + {0..7} and k0 + 8*hi + {16..23}
__device__ __forceinline__ v16h load_a_frag(const _Float16* base, int ld,
                                            int row, int k0, int hi) {
  const _Float16* p = base + (size_t)row * ld + k0 + 8 * hi;
  v8h lo = *(const v8h*)(p);
  v8h hh = *(const v8h*)(p + 16);
  v16h a;
#pragma unroll
  for (int j = 0; j < 8; ++j) { a[j] = lo[j]; a[j + 8] = hh[j]; }
  return a;
}

// B 32x16 (KxN): lane n holds column n; per-lane halfs: K = k0 + 8*hi + {0..7} and k0 + 8*hi + {16..23}
// source is row-major [n][k] (i.e. a weight row == B column)
__device__ __forceinline__ v16h load_b_frag(const _Float16* base, int ld,
                                            int col, int k0, int hi) {
  const _Float16* p = base + (size_t)col * ld + k0 + 8 * hi;
  v8h lo = *(const v8h*)(p);
  v8h hh = *(const v8h*)(p + 16);
  v16h b;
#pragma unroll
  for (int j = 0; j < 8; ++j) { b[j] = lo[j]; b[j + 8] = hh[j]; }
  return b;
}

// ---------------- prep: conversions / padding ------------------------------
__global__ __launch_bounds__(256) void ndf_prep(
    const float* __restrict__ x,  const float* __restrict__ W1,
    const float* __restrict__ W2, const float* __restrict__ Wd,
    const float* __restrict__ bd, const float* __restrict__ ll,
    _Float16* __restrict__ x16, _Float16* __restrict__ W1h,
    _Float16* __restrict__ W2h, _Float16* __restrict__ Wdp,
    float* __restrict__ bdp, float* __restrict__ leafp) {
  int tid = blockIdx.x * blockDim.x + threadIdx.x;
  int stride = gridDim.x * blockDim.x;
  for (int i = tid; i < BATCH * IND; i += stride) x16[i] = (_Float16)x[i];
  for (int i = tid; i < H1DIM * IND; i += stride) W1h[i] = (_Float16)W1[i];
  for (int i = tid; i < H2DIM * H1DIM; i += stride) W2h[i] = (_Float16)W2[i];
  for (int i = tid; i < NTREES * 256 * 64; i += stride) {
    int t = i >> 14;            // / (256*64)
    int node = (i >> 6) & 255;
    int k = i & 63;
    Wdp[i] = (node < NNODES) ? (_Float16)Wd[((size_t)t * NNODES + node) * 64 + k]
                             : (_Float16)0.0f;
  }
  for (int i = tid; i < NTREES * 256; i += stride) {
    int t = i >> 8; int node = i & 255;
    bdp[i] = (node < NNODES) ? bd[t * NNODES + node] : 0.0f;
  }
  for (int i = tid; i < NTREES * NLEAF; i += stride)
    leafp[i] = 1.0f / (1.0f + __expf(-ll[i]));
}

// ---------------- mlp1: h1 = relu(x @ W1^T + b1) ---------------------------
__global__ __launch_bounds__(256) void ndf_mlp1(
    const _Float16* __restrict__ x16, const _Float16* __restrict__ W1h,
    const float* __restrict__ b1, _Float16* __restrict__ h1h) {
  int wave = threadIdx.x >> 5, lane = threadIdx.x & 31;
  int hi = lane >> 4, lr = lane & 15;
  int m0 = blockIdx.x * 16;      // 512 blocks
  int n0 = wave * 16;            // 8 waves -> N = 128
  v8f c = {};
#pragma unroll
  for (int kk = 0; kk < IND; kk += 32) {
    v16h a = load_a_frag(x16 + (size_t)m0 * IND, IND, lr, kk, hi);
    v16h b = load_b_frag(W1h, IND, n0 + lr, kk, hi);
    c = __builtin_amdgcn_wmma_f32_16x16x32_f16(false, a, false, b,
                                               (short)0, c, false, false);
  }
  float bias = b1[n0 + lr];
#pragma unroll
  for (int r = 0; r < 8; ++r) {
    int m = r + 8 * hi;
    float v = c[r] + bias;
    v = v > 0.0f ? v : 0.0f;
    h1h[(size_t)(m0 + m) * H1DIM + n0 + lr] = (_Float16)v;
  }
}

// ---------------- mlp2: feats = relu(h1 @ W2^T + b2) -----------------------
__global__ __launch_bounds__(128) void ndf_mlp2(
    const _Float16* __restrict__ h1h, const _Float16* __restrict__ W2h,
    const float* __restrict__ b2, _Float16* __restrict__ featsH) {
  int wave = threadIdx.x >> 5, lane = threadIdx.x & 31;
  int hi = lane >> 4, lr = lane & 15;
  int m0 = blockIdx.x * 16;      // 512 blocks
  int n0 = wave * 16;            // 4 waves -> N = 64
  v8f c = {};
#pragma unroll
  for (int kk = 0; kk < H1DIM; kk += 32) {
    v16h a = load_a_frag(h1h + (size_t)m0 * H1DIM, H1DIM, lr, kk, hi);
    v16h b = load_b_frag(W2h, H1DIM, n0 + lr, kk, hi);
    c = __builtin_amdgcn_wmma_f32_16x16x32_f16(false, a, false, b,
                                               (short)0, c, false, false);
  }
  float bias = b2[n0 + lr];
#pragma unroll
  for (int r = 0; r < 8; ++r) {
    int m = r + 8 * hi;
    float v = c[r] + bias;
    v = v > 0.0f ? v : 0.0f;
    featsH[(size_t)(m0 + m) * H2DIM + n0 + lr] = (_Float16)v;
  }
}

// ---------------- forest: fused GEMM3 + sigmoid + routing ------------------
// grid = (256 batch tiles of 32, 8 tree-groups); block = 8 waves, 1 tree/wave.
// Output tiles are (node x batch): lane column = batch row, accumulator
// registers r=0..7 are nodes nt*16 + r + 8*hi. Two batch sub-tiles per wave
// share each Wd A-fragment (doubles FLOPs per Wd byte).
__global__ __launch_bounds__(256) void ndf_forest(
    const _Float16* __restrict__ featsH, const _Float16* __restrict__ Wdp,
    const float* __restrict__ bdp, const float* __restrict__ leafp,
    float* __restrict__ part) {
  __shared__ float sLeaf[8 * 256];
  __shared__ float sBd[8 * 256];
  __shared__ float sPart[8][32];

  int wave = threadIdx.x >> 5, lane = threadIdx.x & 31;
  int hi = lane >> 4, lr = lane & 15;
  int b0 = blockIdx.x * 32;
  int g  = blockIdx.y;
  int t  = g * 8 + wave;

#pragma unroll
  for (int i = 0; i < 8; ++i) {
    sLeaf[wave * 256 + lane + i * 32] = leafp[(size_t)t * 256 + lane + i * 32];
    sBd[wave * 256 + lane + i * 32]   = bdp[(size_t)t * 256 + lane + i * 32];
  }

  const _Float16* Abase = Wdp + (size_t)t * 256 * 64;
  // Pull this tree's 32KB Wd slice toward the caches (global_prefetch_b8).
#pragma unroll
  for (int i = 0; i < 8; ++i)
    __builtin_prefetch(Abase + (size_t)(i * 32 + lane) * 64, 0, 3);

  __syncthreads();

  // B fragments: feats^T tiles (K=64 x batch 16) x 2 sub-tiles; reused for
  // all 16 node tiles of the tree.
  v16h bf[2][2];
#pragma unroll
  for (int s2 = 0; s2 < 2; ++s2) {
    bf[s2][0] = load_b_frag(featsH + (size_t)(b0 + s2 * 16) * H2DIM, H2DIM, lr, 0, hi);
    bf[s2][1] = load_b_frag(featsH + (size_t)(b0 + s2 * 16) * H2DIM, H2DIM, lr, 32, hi);
  }

  float sP[2][8], qP[2][8];
#pragma unroll
  for (int s2 = 0; s2 < 2; ++s2)
#pragma unroll
    for (int l = 0; l < 8; ++l) { sP[s2][l] = 1.0f; qP[s2][l] = 1.0f; }

#pragma unroll
  for (int nt = 0; nt < 16; ++nt) {
    v16h a0 = load_a_frag(Abase + nt * 16 * 64, 64, lr, 0, hi);
    v16h a1 = load_a_frag(Abase + nt * 16 * 64, 64, lr, 32, hi);
    v8f c0 = {};
    c0 = __builtin_amdgcn_wmma_f32_16x16x32_f16(false, a0, false, bf[0][0],
                                                (short)0, c0, false, false);
    c0 = __builtin_amdgcn_wmma_f32_16x16x32_f16(false, a1, false, bf[0][1],
                                                (short)0, c0, false, false);
    v8f c1 = {};
    c1 = __builtin_amdgcn_wmma_f32_16x16x32_f16(false, a0, false, bf[1][0],
                                                (short)0, c1, false, false);
    c1 = __builtin_amdgcn_wmma_f32_16x16x32_f16(false, a1, false, bf[1][1],
                                                (short)0, c1, false, false);
#pragma unroll
    for (int r = 0; r < 8; ++r) {
      const int mA = nt * 16 + r;          // node if hi==0 (compile-time)
      const int mB = mA + 8;               // node if hi==1 (compile-time)
      const int lvA = 31 - __builtin_clz(mA + 1);  // constant-folds
      const int lvB = 31 - __builtin_clz(mB + 1);  // constant-folds
      float bias = sBd[wave * 256 + (hi ? mB : mA)];
#pragma unroll
      for (int s2 = 0; s2 < 2; ++s2) {
        float z  = (s2 == 0 ? c0[r] : c1[r]) + bias;
        float dv = 1.0f / (1.0f + __expf(-z));
        float nd = 1.0f - dv;
        if (!hi) {                     // mA <= 247 < 255 always valid
          sP[s2][lvA] *= dv; qP[s2][lvA] *= nd;
        } else if (mB < 255) {         // skip padded node 255
          sP[s2][lvB] *= dv; qP[s2][lvB] *= nd;
        }
      }
    }
  }

  // Lanes 0-15 hold products over nodes with bit3==0, lanes 16-31 the rest.
  // Combine halves, then contract the 256 leaves (factored 4+4 levels).
  const float* Lp = &sLeaf[wave * 256];
  float treeOut[2];
#pragma unroll
  for (int s2 = 0; s2 < 2; ++s2) {
    float s[8], q[8];
#pragma unroll
    for (int l = 0; l < 8; ++l) {
      s[l] = sP[s2][l] * __shfl_xor(sP[s2][l], 16, 32);
      q[l] = qP[s2][l] * __shfl_xor(qP[s2][l], 16, 32);
    }
    float Bv[16];
#pragma unroll
    for (int lo = 0; lo < 16; ++lo) {
      float v = ((lo >> 3) & 1) ? s[4] : q[4];
      v *= ((lo >> 2) & 1) ? s[5] : q[5];
      v *= ((lo >> 1) & 1) ? s[6] : q[6];
      v *= ((lo >> 0) & 1) ? s[7] : q[7];
      Bv[lo] = v;
    }
    float acc = 0.0f;
#pragma unroll
    for (int hi4 = 0; hi4 < 16; ++hi4) {
      float Av = (((hi4 >> 3) & 1) ? s[0] : q[0]) *
                 (((hi4 >> 2) & 1) ? s[1] : q[1]) *
                 (((hi4 >> 1) & 1) ? s[2] : q[2]) *
                 (((hi4 >> 0) & 1) ? s[3] : q[3]);
      float inner = 0.0f;
#pragma unroll
      for (int lo = 0; lo < 16; ++lo) inner += Bv[lo] * Lp[hi4 * 16 + lo];
      acc += Av * inner;
    }
    treeOut[s2] = acc * (1.0f / 256.0f);   // initial leaf_probs = 1/256
  }

  if (hi == 0) {                           // halves are redundant copies
    sPart[wave][lr]      = treeOut[0];
    sPart[wave][16 + lr] = treeOut[1];
  }
  __syncthreads();
  if (threadIdx.x < 32) {
    float sum = 0.0f;
#pragma unroll
    for (int w = 0; w < 8; ++w) sum += sPart[w][threadIdx.x];
    part[(size_t)g * BATCH + b0 + threadIdx.x] = sum;
  }
}

// ---------------- final mean over trees ------------------------------------
__global__ __launch_bounds__(256) void ndf_reduce(const float* __restrict__ part,
                                                  float* __restrict__ out) {
  int b = blockIdx.x * blockDim.x + threadIdx.x;
  if (b < BATCH) {
    float sum = 0.0f;
#pragma unroll
    for (int gg = 0; gg < 8; ++gg) sum += part[(size_t)gg * BATCH + b];
    out[b] = sum * (1.0f / (float)NTREES);
  }
}

// ---------------- host launch ----------------------------------------------
extern "C" void kernel_launch(void* const* d_in, const int* in_sizes, int n_in,
                              void* d_out, int out_size, void* d_ws, size_t ws_size,
                              hipStream_t stream) {
  const float* x  = (const float*)d_in[0];
  const float* W1 = (const float*)d_in[1];
  const float* b1 = (const float*)d_in[2];
  const float* W2 = (const float*)d_in[3];
  const float* b2 = (const float*)d_in[4];
  const float* Wd = (const float*)d_in[5];
  const float* bd = (const float*)d_in[6];
  const float* ll = (const float*)d_in[7];
  float* out = (float*)d_out;

  char* ws = (char*)d_ws;
  size_t off = 0;
  _Float16* x16    = (_Float16*)(ws + off); off += (size_t)BATCH * IND * 2;        // 4 MB
  _Float16* W1h    = (_Float16*)(ws + off); off += (size_t)H1DIM * IND * 2;        // 64 KB
  _Float16* W2h    = (_Float16*)(ws + off); off += (size_t)H2DIM * H1DIM * 2;      // 16 KB
  _Float16* h1h    = (_Float16*)(ws + off); off += (size_t)BATCH * H1DIM * 2;      // 2 MB
  _Float16* featsH = (_Float16*)(ws + off); off += (size_t)BATCH * H2DIM * 2;      // 1 MB
  _Float16* Wdp    = (_Float16*)(ws + off); off += (size_t)NTREES * 256 * 64 * 2;  // 2 MB
  float*    bdp    = (float*)(ws + off);    off += (size_t)NTREES * 256 * 4;       // 64 KB
  float*    leafp  = (float*)(ws + off);    off += (size_t)NTREES * 256 * 4;       // 64 KB
  float*    part   = (float*)(ws + off);    off += (size_t)8 * BATCH * 4;          // 256 KB
  (void)ws_size; (void)in_sizes; (void)n_in; (void)out_size;

  ndf_prep<<<2048, 256, 0, stream>>>(x, W1, W2, Wd, bd, ll,
                                     x16, W1h, W2h, Wdp, bdp, leafp);
  ndf_mlp1<<<BATCH / 16, 256, 0, stream>>>(x16, W1h, b1, h1h);
  ndf_mlp2<<<BATCH / 16, 128, 0, stream>>>(h1h, W2h, b2, featsH);
  ndf_forest<<<dim3(BATCH / 32, 8), 256, 0, stream>>>(featsH, Wdp, bdp, leafp, part);
  ndf_reduce<<<(BATCH + 255) / 256, 256, 0, stream>>>(part, out);
}
